// GeometryAwareAttention_84413287236148
// MI455X (gfx1250) — compile-verified, matrix operand order checked by reading, not run on hardware
//
#include <hip/hip_runtime.h>

typedef __attribute__((ext_vector_type(16))) _Float16 v16h;
typedef __attribute__((ext_vector_type(8)))  _Float16 h8;
typedef __attribute__((ext_vector_type(4)))  _Float16 h4;
typedef __attribute__((ext_vector_type(8)))  float    v8f;

#define B_   4
#define H_   8
#define M_   2048
#define D_   512
#define DK_  64

static __device__ __forceinline__ v16h combine16(h8 lo, h8 hi) {
  v16h r;
#pragma unroll
  for (int i = 0; i < 8; ++i) { r[i] = lo[i]; r[i + 8] = hi[i]; }
  return r;
}

// one lane copies 16 bytes global -> LDS asynchronously (ASYNCcnt-tracked)
static __device__ __forceinline__ void async_copy_b128(const _Float16* g,
                                                       _Float16* l) {
  const unsigned lds_off = (unsigned)(unsigned long long)l; // LDS aperture: low 32 bits = LDS byte offset
  asm volatile("global_load_async_to_lds_b128 %0, %1, off"
               :: "v"(lds_off), "v"(g) : "memory");
}

// ---------------------------------------------------------------------------
// Kernel 1: QKV projection  y[r][e] = sum_d x[r][d] * W[e][d] + bias[e]
//   Q,K stored as f16 [B*H][M][64]; V stored transposed [B*H][64][M].
// ---------------------------------------------------------------------------
__global__ __launch_bounds__(256) void qkv_proj_kernel(
    const float* __restrict__ x,
    const float* __restrict__ wq, const float* __restrict__ bq,
    const float* __restrict__ wk, const float* __restrict__ bk,
    const float* __restrict__ wv, const float* __restrict__ bv,
    _Float16* __restrict__ Q, _Float16* __restrict__ K, _Float16* __restrict__ Vt)
{
  const int mat = blockIdx.z;
  const float* W  = (mat == 0) ? wq : (mat == 1) ? wk : wv;
  const float* bb = (mat == 0) ? bq : (mat == 1) ? bk : bv;

  const int tid  = threadIdx.x;
  const int wid  = tid >> 5;
  const int lane = tid & 31;
  const int ln16 = lane & 15;
  const int g    = lane >> 4;

  const int rbase = blockIdx.x * 16;              // rows over B*M = 8192
  const int e0    = blockIdx.y * 128 + wid * 16;  // output column tile

  __shared__ __align__(16) _Float16 xt[16 * 520];

  for (int i = tid; i < 16 * 128; i += 256) {
    const int row = i >> 7, c4 = i & 127;
    const float4 v = *reinterpret_cast<const float4*>(
        x + (size_t)(rbase + row) * D_ + c4 * 4);
    h4 hv; hv[0] = (_Float16)v.x; hv[1] = (_Float16)v.y;
    hv[2] = (_Float16)v.z; hv[3] = (_Float16)v.w;
    *reinterpret_cast<h4*>(&xt[row * 520 + c4 * 4]) = hv;
  }
  __syncthreads();

  v8f acc = {};
#pragma unroll 4
  for (int kt = 0; kt < 16; ++kt) {
    const int k0 = kt * 32;
    const h8 alo = *reinterpret_cast<const h8*>(&xt[ln16 * 520 + k0 + g * 8]);
    const h8 ahi = *reinterpret_cast<const h8*>(&xt[ln16 * 520 + k0 + 16 + g * 8]);
    const v16h a = combine16(alo, ahi);
    const float* wr = W + (size_t)(e0 + ln16) * D_ + k0 + g * 8;
    const float4 w0 = *reinterpret_cast<const float4*>(wr);
    const float4 w1 = *reinterpret_cast<const float4*>(wr + 4);
    const float4 w2 = *reinterpret_cast<const float4*>(wr + 16);
    const float4 w3 = *reinterpret_cast<const float4*>(wr + 20);
    v16h bf;
    bf[0]  = (_Float16)w0.x; bf[1]  = (_Float16)w0.y;
    bf[2]  = (_Float16)w0.z; bf[3]  = (_Float16)w0.w;
    bf[4]  = (_Float16)w1.x; bf[5]  = (_Float16)w1.y;
    bf[6]  = (_Float16)w1.z; bf[7]  = (_Float16)w1.w;
    bf[8]  = (_Float16)w2.x; bf[9]  = (_Float16)w2.y;
    bf[10] = (_Float16)w2.z; bf[11] = (_Float16)w2.w;
    bf[12] = (_Float16)w3.x; bf[13] = (_Float16)w3.y;
    bf[14] = (_Float16)w3.z; bf[15] = (_Float16)w3.w;
    acc = __builtin_amdgcn_wmma_f32_16x16x32_f16(false, a, false, bf,
                                                 (short)0, acc, false, false);
  }

  const int   e  = e0 + ln16;
  const float be = bb[e];
  const int   hh = e >> 6, dk = e & 63;
#pragma unroll
  for (int v = 0; v < 8; ++v) {
    const int r  = rbase + v + 8 * g;
    const int bi = r >> 11, m = r & (M_ - 1);
    const _Float16 val = (_Float16)(acc[v] + be);
    if (mat == 0)
      Q[(size_t)((bi * H_ + hh) * M_ + m) * DK_ + dk] = val;
    else if (mat == 1)
      K[(size_t)((bi * H_ + hh) * M_ + m) * DK_ + dk] = val;
    else
      Vt[(size_t)((bi * H_ + hh) * DK_ + dk) * M_ + m] = val;
  }
}

// ---------------------------------------------------------------------------
// Kernel 2: fused flash attention with geometric bias.
//   K/V chunks staged into LDS with double-buffered async copies.
//   grid = (M/128, B*H), block = 256; each wave owns 16 query rows.
// ---------------------------------------------------------------------------
__global__ __launch_bounds__(256) void attn_kernel(
    const _Float16* __restrict__ Q, const _Float16* __restrict__ K,
    const _Float16* __restrict__ Vt, const float* __restrict__ coords,
    const float* __restrict__ alpha_p, float* __restrict__ out)
{
  const int tid  = threadIdx.x;
  const int wid  = tid >> 5;
  const int lane = tid & 31;
  const int ln16 = lane & 15;
  const int g    = lane >> 4;
  const int bh   = blockIdx.y, b = bh >> 3, h = bh & 7;
  const int qbase = blockIdx.x * 128 + wid * 16;
  const float alpha = alpha_p[0];

  const _Float16* qp = Q  + (size_t)bh * M_ * DK_;
  const _Float16* kp = K  + (size_t)bh * M_ * DK_;
  const _Float16* vp = Vt + (size_t)bh * DK_ * M_;
  const float*    cb = coords + (size_t)b * M_ * 3;

  __shared__ __align__(16) _Float16 kbuf[2][32 * DK_];  // 2 x 4 KB
  __shared__ __align__(16) _Float16 vbuf[2][DK_ * 32];  // 2 x 4 KB
  __shared__ __align__(16) _Float16 pbuf[8][16 * 40];   // per-wave P staging
  _Float16* pb = &pbuf[wid][0];

  // Q A-fragments for the two d-halves (DK=64 -> 2 x K=32)
  v16h aq[2];
#pragma unroll
  for (int dt = 0; dt < 2; ++dt) {
    const _Float16* p = qp + (size_t)(qbase + ln16) * DK_ + dt * 32 + g * 8;
    aq[dt] = combine16(*reinterpret_cast<const h8*>(p),
                       *reinterpret_cast<const h8*>(p + 16));
  }

  // loop-invariant coords of this wave's 16 query rows
  float cmx[8], cmy[8], cmz[8];
#pragma unroll
  for (int v = 0; v < 8; ++v) {
    const int m = qbase + v + 8 * g;
    cmx[v] = cb[m * 3 + 0]; cmy[v] = cb[m * 3 + 1]; cmz[v] = cb[m * 3 + 2];
  }

  float rmax[8], rsum[8];
#pragma unroll
  for (int v = 0; v < 8; ++v) { rmax[v] = -3.0e38f; rsum[v] = 0.0f; }
  v8f o0 = {}, o1 = {}, o2 = {}, o3 = {};

  const int vr = tid >> 2, vq = tid & 3; // V-chunk copy assignment (64 rows x 4x16B)

  // prologue: async-stage chunk 0 into buffer 0 (2 x b128 per thread = 8 KB)
  async_copy_b128(kp + (size_t)0 * DK_ + tid * 8, &kbuf[0][tid * 8]);
  async_copy_b128(vp + (size_t)vr * M_ + 0 + vq * 8, &vbuf[0][vr * 32 + vq * 8]);

  for (int nb = 0; nb < M_ / 32; ++nb) {
    const int cur = nb & 1;
    const int nbase = nb * 32;

    asm volatile("s_wait_asynccnt 0x0" ::: "memory"); // chunk nb landed in LDS
    __syncthreads();                                  // visible to all waves;
                                                      // all done reading buf cur^1
    if (nb + 1 < M_ / 32) {                           // prefetch chunk nb+1
      const int nn = nbase + 32;
      async_copy_b128(kp + (size_t)nn * DK_ + tid * 8, &kbuf[cur ^ 1][tid * 8]);
      async_copy_b128(vp + (size_t)vr * M_ + nn + vq * 8,
                      &vbuf[cur ^ 1][vr * 32 + vq * 8]);
    }

    const _Float16* kl = &kbuf[cur][0];
    const _Float16* vl = &vbuf[cur][0];

    // --- S = Q * K^T on a 16x32 chunk (two 16x16 output tiles) ---
    v8f s0 = {}, s1 = {};
#pragma unroll
    for (int dt = 0; dt < 2; ++dt) {
      const _Float16* p0 = kl + ln16 * DK_ + dt * 32 + g * 8;
      const v16h bk0 = combine16(*reinterpret_cast<const h8*>(p0),
                                 *reinterpret_cast<const h8*>(p0 + 16));
      s0 = __builtin_amdgcn_wmma_f32_16x16x32_f16(false, aq[dt], false, bk0,
                                                  (short)0, s0, false, false);
      const _Float16* p1 = kl + (16 + ln16) * DK_ + dt * 32 + g * 8;
      const v16h bk1 = combine16(*reinterpret_cast<const h8*>(p1),
                                 *reinterpret_cast<const h8*>(p1 + 16));
      s1 = __builtin_amdgcn_wmma_f32_16x16x32_f16(false, aq[dt], false, bk1,
                                                  (short)0, s1, false, false);
    }

    // geometric bias columns for this chunk
    const int   n0 = nbase + ln16, n1 = n0 + 16;
    const float cn0x = cb[n0 * 3], cn0y = cb[n0 * 3 + 1], cn0z = cb[n0 * 3 + 2];
    const float cn1x = cb[n1 * 3], cn1y = cb[n1 * 3 + 1], cn1z = cb[n1 * 3 + 2];

    // --- online softmax (row = half-wave of 16 lanes, same VGPR) ---
#pragma unroll
    for (int v = 0; v < 8; ++v) {
      const float a0 = s0[v] * 0.125f +
                       alpha * (cmx[v] * cn0x + cmy[v] * cn0y + cmz[v] * cn0z);
      const float a1 = s1[v] * 0.125f +
                       alpha * (cmx[v] * cn1x + cmy[v] * cn1y + cmz[v] * cn1z);
      float mx = fmaxf(a0, a1);
#pragma unroll
      for (int off = 8; off >= 1; off >>= 1)
        mx = fmaxf(mx, __shfl_xor(mx, off, 32));
      const float nm  = fmaxf(rmax[v], mx);
      const float sc  = __expf(rmax[v] - nm);
      const float p0v = __expf(a0 - nm);
      const float p1v = __expf(a1 - nm);
      float ps = p0v + p1v;
#pragma unroll
      for (int off = 8; off >= 1; off >>= 1)
        ps += __shfl_xor(ps, off, 32);
      rsum[v] = rsum[v] * sc + ps;
      rmax[v] = nm;
      o0[v] *= sc; o1[v] *= sc; o2[v] *= sc; o3[v] *= sc;
      pb[(v + 8 * g) * 40 + ln16]      = (_Float16)p0v;
      pb[(v + 8 * g) * 40 + 16 + ln16] = (_Float16)p1v;
    }
    asm volatile("s_wait_dscnt 0x0" ::: "memory");

    // re-read P as a 16x32 A-fragment
    const v16h ap = combine16(
        *reinterpret_cast<const h8*>(&pb[ln16 * 40 + g * 8]),
        *reinterpret_cast<const h8*>(&pb[ln16 * 40 + 16 + g * 8]));

    // --- O += P * V (four 16x16 d-column tiles) ---
#pragma unroll
    for (int dt = 0; dt < 4; ++dt) {
      const _Float16* pv = vl + (dt * 16 + ln16) * 32 + g * 8;
      const v16h bv = combine16(*reinterpret_cast<const h8*>(pv),
                                *reinterpret_cast<const h8*>(pv + 16));
      v8f* op = (dt == 0) ? &o0 : (dt == 1) ? &o1 : (dt == 2) ? &o2 : &o3;
      *op = __builtin_amdgcn_wmma_f32_16x16x32_f16(false, ap, false, bv,
                                                   (short)0, *op, false, false);
    }
  }

  // epilogue: normalize and scatter to [B][M][512]
#pragma unroll
  for (int dt = 0; dt < 4; ++dt) {
    const v8f& o = (dt == 0) ? o0 : (dt == 1) ? o1 : (dt == 2) ? o2 : o3;
#pragma unroll
    for (int v = 0; v < 8; ++v) {
      const int m = qbase + v + 8 * g;
      const int e = h * DK_ + dt * 16 + ln16;
      out[(size_t)(b * M_ + m) * D_ + e] = o[v] / rsum[v];
    }
  }
}

// ---------------------------------------------------------------------------
extern "C" void kernel_launch(void* const* d_in, const int* in_sizes, int n_in,
                              void* d_out, int out_size, void* d_ws, size_t ws_size,
                              hipStream_t stream) {
  const float* x      = (const float*)d_in[0];
  const float* coords = (const float*)d_in[1];
  const float* wq     = (const float*)d_in[2];
  const float* bq     = (const float*)d_in[3];
  const float* wk     = (const float*)d_in[4];
  const float* bk     = (const float*)d_in[5];
  const float* wv     = (const float*)d_in[6];
  const float* bv     = (const float*)d_in[7];
  const float* alpha  = (const float*)d_in[8];
  float* out = (float*)d_out;

  _Float16* ws = (_Float16*)d_ws;
  const size_t QSZ = (size_t)B_ * H_ * M_ * DK_;   // 4,194,304 halves = 8 MB
  _Float16* Qb = ws;
  _Float16* Kb = ws + QSZ;
  _Float16* Vb = ws + 2 * QSZ;                     // total 24 MB of d_ws

  dim3 g1((B_ * M_) / 16, D_ / 128, 3);
  qkv_proj_kernel<<<g1, 256, 0, stream>>>(x, wq, bq, wk, bk, wv, bv, Qb, Kb, Vb);

  dim3 g2(M_ / 128, B_ * H_);
  attn_kernel<<<g2, 256, 0, stream>>>(Qb, Kb, Vb, coords, alpha, out);
}
